// VectorQuantizerEMA_13872744366251
// MI455X (gfx1250) — compile-verified, matrix operand order checked by reading, not run on hardware
//
#include <hip/hip_runtime.h>
#include <hip/hip_bf16.h>

#define NUM_CODES 512
#define CODE_DIM  64
#define N_ROWS    (64 * 4096)            // 262144
#define N_ELEMS   (N_ROWS * CODE_DIM)    // 16777216
#define EMA_DECAY 0.99f
#define ONE_MINUS 0.01f
#define COMMIT    0.25f
#define EPSV      1e-6f

typedef __attribute__((ext_vector_type(16))) __bf16 v16bf;
typedef __attribute__((ext_vector_type(8)))  float  v8f;
typedef __attribute__((vector_size(16)))     int    vs4i;   // async builtin elem type

// workspace layout (bytes)
#define WS_CODES   0          // 512*64 bf16  = 65536 B
#define WS_NORMS   65536      // 512 f32      = 2048 B
#define WS_COUNTS  67584      // 512 f32      = 2048 B
#define WS_ESUM    69632      // 512*64 f32   = 131072 B
#define WS_LOSS    200704     // 1 f32

#define AS3 __attribute__((address_space(3)))
#define AS1 __attribute__((address_space(1)))

// Async global->LDS 16B copy (ASYNCcnt path) with safe fallback.
__device__ static inline void async_copy16(void* lds_dst, const void* gsrc)
{
#if __has_builtin(__builtin_amdgcn_global_load_async_to_lds_b128)
    __builtin_amdgcn_global_load_async_to_lds_b128(
        (AS1 vs4i*)const_cast<void*>(gsrc), (AS3 vs4i*)lds_dst, 0, 0);
#else
    *(uint4*)lds_dst = *(const uint4*)gsrc;
#endif
}

__device__ static inline void wait_async_all()
{
#if __has_builtin(__builtin_amdgcn_s_wait_asynccnt)
    __builtin_amdgcn_s_wait_asynccnt(0);
#else
    asm volatile("s_wait_asynccnt 0" ::: "memory");
#endif
}

// ---------------------------------------------------------------------------
// Kernel 0: cast codebook to bf16, compute ||e||^2, zero accumulators.
// grid = 512 blocks (one per code), block = 64 threads (one per dim)
// ---------------------------------------------------------------------------
__global__ void vq_prep(const float* __restrict__ embedding,
                        unsigned char* __restrict__ ws)
{
    __bf16* codes = (__bf16*)(ws + WS_CODES);
    float*  norms = (float*)(ws + WS_NORMS);
    float*  cnts  = (float*)(ws + WS_COUNTS);
    float*  esum  = (float*)(ws + WS_ESUM);
    float*  lossw = (float*)(ws + WS_LOSS);

    const int c = blockIdx.x;   // code
    const int d = threadIdx.x;  // dim

    const float e = embedding[c * CODE_DIM + d];
    codes[c * CODE_DIM + d] = (__bf16)e;
    esum[c * CODE_DIM + d]  = 0.0f;

    __shared__ float s[CODE_DIM];
    s[d] = e * e;
    __syncthreads();
    if (d == 0) {
        float acc = 0.0f;
        #pragma unroll
        for (int i = 0; i < CODE_DIM; ++i) acc += s[i];
        norms[c] = acc;
        cnts[c]  = 0.0f;
        if (c == 0) *lossw = 0.0f;
    }
}

// ---------------------------------------------------------------------------
// Kernel 1: WMMA distance GEMM + argmin + gather + segment-sum atomics.
// grid = 512 blocks x 256 threads (8 waves). Each wave: 4 row-tiles of 16.
// Codebook (bf16) + norms staged in LDS via async global->LDS copies.
// ---------------------------------------------------------------------------
__global__ __launch_bounds__(256) void vq_main(
    const float* __restrict__ z,
    const float* __restrict__ embedding,
    unsigned char* __restrict__ ws,
    float* __restrict__ out)
{
    float* cnts  = (float*)(ws + WS_COUNTS);
    float* esum  = (float*)(ws + WS_ESUM);
    float* lossw = (float*)(ws + WS_LOSS);

    float* out_zq  = out;                      // [N_ROWS*64] z_q (straight-through == z_q)
    float* out_idx = out + N_ELEMS + 1;        // [N_ROWS] indices as float

    __shared__ __align__(16) __bf16 sCodes[NUM_CODES * CODE_DIM]; // 64 KB
    __shared__ __align__(16) float  sNorms[NUM_CODES];            // 2 KB
    __shared__ int sIdx[8][16];                                    // per-wave winners

    const int tid = threadIdx.x;
    const int w   = tid >> 5;    // wave in block
    const int ln  = tid & 31;    // lane
    const int lnM = ln & 15;
    const int h   = ln >> 4;     // lane half selects K-subgroup

    // ---- stage codebook + norms into LDS (async DMA, 16B granules) ----
    {
        const unsigned char* gcodes = ws + WS_CODES;
        unsigned char* lcodes = (unsigned char*)sCodes;
        #pragma unroll
        for (int i = 0; i < 16; ++i) {           // 256 thr * 16 * 16B = 64 KB
            const int byt = (tid + i * 256) * 16;
            async_copy16(lcodes + byt, gcodes + byt);
        }
        if (tid < 128) {                         // 128 * 16B = 2 KB norms
            const int byt = tid * 16;
            async_copy16((unsigned char*)sNorms + byt, ws + WS_NORMS + byt);
        }
        wait_async_all();
        __syncthreads();
    }

    const int waveGlobal = blockIdx.x * 8 + w;   // 0..4095
    float lossAcc = 0.0f;

    for (int t = 0; t < 4; ++t) {
        const int tile    = waveGlobal + t * 4096;  // 0..16383
        const int rowBase = tile * 16;
        const int myRow   = rowBase + lnM;

        // ---- A fragments (16x32 bf16 each): a0 = K[0..31], a1 = K[32..63] ----
        v16bf a0, a1;
        {
            const float* zr = z + (size_t)myRow * CODE_DIM;
            #pragma unroll
            for (int f = 0; f < 2; ++f) {
                const int k0 = f * 32 + h * 8;
                float4 p0 = *(const float4*)(zr + k0);
                float4 p1 = *(const float4*)(zr + k0 + 4);
                float4 p2 = *(const float4*)(zr + k0 + 16);
                float4 p3 = *(const float4*)(zr + k0 + 20);
                v16bf& a = f ? a1 : a0;
                a[0]=(__bf16)p0.x;  a[1]=(__bf16)p0.y;  a[2]=(__bf16)p0.z;  a[3]=(__bf16)p0.w;
                a[4]=(__bf16)p1.x;  a[5]=(__bf16)p1.y;  a[6]=(__bf16)p1.z;  a[7]=(__bf16)p1.w;
                a[8]=(__bf16)p2.x;  a[9]=(__bf16)p2.y;  a[10]=(__bf16)p2.z; a[11]=(__bf16)p2.w;
                a[12]=(__bf16)p3.x; a[13]=(__bf16)p3.y; a[14]=(__bf16)p3.z; a[15]=(__bf16)p3.w;
            }
        }

        float bestd[8];
        int   besti[8];
        #pragma unroll
        for (int r = 0; r < 8; ++r) { bestd[r] = 3.4e38f; besti[r] = 0; }

        // ---- sweep all 512 codes in 16-code tiles (B from LDS) ----
        for (int ct = 0; ct < 32; ++ct) {
            const int n = ct * 16 + lnM;             // my B column (code id)
            union { v16bf v; uint4 u[2]; } b0, b1;
            const __bf16* cr = sCodes + n * CODE_DIM; // row-major bf16 => fragment order
            b0.u[0] = *(const uint4*)(cr + h * 8);
            b0.u[1] = *(const uint4*)(cr + h * 8 + 16);
            b1.u[0] = *(const uint4*)(cr + 32 + h * 8);
            b1.u[1] = *(const uint4*)(cr + 32 + h * 8 + 16);
            const float nrm = sNorms[n];

            v8f c = {};
            c = __builtin_amdgcn_wmma_f32_16x16x32_bf16(false, a0, false, b0.v,
                                                        (short)0, c, false, false);
            c = __builtin_amdgcn_wmma_f32_16x16x32_bf16(false, a1, false, b1.v,
                                                        (short)0, c, false, false);

            // C layout: VGPR r -> row (h ? r+8 : r), lane -> code n
            #pragma unroll
            for (int r = 0; r < 8; ++r) {
                float d = nrm - 2.0f * c[r];
                if (d < bestd[r]) { bestd[r] = d; besti[r] = n; }
            }
        }

        // ---- branchless argmin reduce: 64-bit sortable key, umin butterfly ----
        #pragma unroll
        for (int r = 0; r < 8; ++r) {
            unsigned u = __float_as_uint(bestd[r]);
            u = (u & 0x80000000u) ? ~u : (u | 0x80000000u);   // ordered-uint map
            unsigned long long key =
                ((unsigned long long)u << 32) | (unsigned)besti[r];
            #pragma unroll
            for (int off = 1; off < 16; off <<= 1) {
                unsigned long long o = __shfl_xor(key, off, 16);
                key = (o < key) ? o : key;
            }
            besti[r] = (int)(unsigned)key;   // smallest dist, tie -> smallest idx
        }

        // stage per-row winners to LDS (lane 0 -> rows 0..7, lane 16 -> rows 8..15)
        #pragma unroll
        for (int r = 0; r < 8; ++r) {
            if (ln == 0)  sIdx[w][r]     = besti[r];
            if (ln == 16) sIdx[w][8 + r] = besti[r];
        }
        asm volatile("s_wait_dscnt 0" ::: "memory");

        // ---- gather z_q, loss, indices, EMA segment sums ----
        #pragma unroll 4
        for (int i = 0; i < 32; ++i) {
            const int e   = i * 32 + ln;   // 0..1023 over the 16x64 tile
            const int mr  = e >> 6;
            const int d   = e & 63;
            const int row = rowBase + mr;
            const int idx = sIdx[w][mr];
            const float zv = z[(size_t)row * CODE_DIM + d];
            const float zq = embedding[idx * CODE_DIM + d];
            out_zq[(size_t)row * CODE_DIM + d] = zq;   // z + sg(z_q - z) == z_q
            const float df = zq - zv;
            lossAcc += df * df;
            atomicAdd(&esum[idx * CODE_DIM + d], zv);
            if (d == 0) {
                atomicAdd(&cnts[idx], 1.0f);
                out_idx[row] = (float)idx;
            }
        }
    }

    // ---- commitment-loss partial: wave reduce, one atomic per wave ----
    #pragma unroll
    for (int off = 1; off < 32; off <<= 1)
        lossAcc += __shfl_xor(lossAcc, off, 32);
    if (ln == 0) atomicAdd(lossw, lossAcc);
}

// ---------------------------------------------------------------------------
// Kernel 2: EMA update + normalization + final loss. 1 block x 512 threads.
// ---------------------------------------------------------------------------
__global__ void vq_final(const float* __restrict__ cluster_size,
                         const float* __restrict__ embedding_avg,
                         const unsigned char* __restrict__ ws,
                         float* __restrict__ out)
{
    const float* cnts  = (const float*)(ws + WS_COUNTS);
    const float* esum  = (const float*)(ws + WS_ESUM);
    const float* lossw = (const float*)(ws + WS_LOSS);

    float* out_loss = out + N_ELEMS;
    float* out_emb  = out + N_ELEMS + 1 + N_ROWS;
    float* out_cs   = out_emb + NUM_CODES * CODE_DIM;
    float* out_avg  = out_cs + NUM_CODES;

    const int t = threadIdx.x;  // code id, 0..511
    const float ncs = cluster_size[t] * EMA_DECAY + ONE_MINUS * cnts[t];
    out_cs[t] = ncs;

    __shared__ float red[NUM_CODES];
    red[t] = ncs;
    __syncthreads();
    for (int s = NUM_CODES / 2; s > 0; s >>= 1) {
        if (t < s) red[t] += red[t + s];
        __syncthreads();
    }
    const float n   = red[0];
    const float cs  = (ncs + EPSV) / (n + NUM_CODES * EPSV);
    const float inv = 1.0f / cs;

    #pragma unroll 8
    for (int d = 0; d < CODE_DIM; ++d) {
        const float nea = embedding_avg[t * CODE_DIM + d] * EMA_DECAY
                        + ONE_MINUS * esum[t * CODE_DIM + d];
        out_avg[t * CODE_DIM + d] = nea;
        out_emb[t * CODE_DIM + d] = nea * inv;
    }
    if (t == 0)
        *out_loss = COMMIT * (*lossw) / (float)N_ELEMS;
}

// ---------------------------------------------------------------------------
extern "C" void kernel_launch(void* const* d_in, const int* in_sizes, int n_in,
                              void* d_out, int out_size, void* d_ws, size_t ws_size,
                              hipStream_t stream)
{
    const float* z   = (const float*)d_in[0];  // [64,4096,64]
    const float* emb = (const float*)d_in[1];  // [512,64]
    const float* csz = (const float*)d_in[2];  // [512]
    const float* eav = (const float*)d_in[3];  // [512,64]
    float* out = (float*)d_out;
    unsigned char* ws = (unsigned char*)d_ws;

    vq_prep <<<NUM_CODES, CODE_DIM, 0, stream>>>(emb, ws);
    vq_main <<<512, 256, 0, stream>>>(z, emb, ws, out);
    vq_final<<<1, NUM_CODES, 0, stream>>>(csz, eav, ws, out);
}
